// EncoderLayer_45629732552839
// MI455X (gfx1250) — matrix operand order repaired and checked by reading, not compiled, not run on hardware
//
#include <hip/hip_runtime.h>
#include <hip/hip_bf16.h>

// ---------------------------------------------------------------------------
// Transformer encoder layer for MI455X (gfx1250, wave32, WMMA).
// B=16, S=1024, D=512, H=8, DH=64, DFF=2048, fp32 I/O, f16 WMMA compute.
// Pre-pass converts activations/weights to f16 (weights transposed to [N][K])
// so GEMM hot loops are pure b128 loads + v_wmma.
// ---------------------------------------------------------------------------

#define B_  16
#define S_  1024
#define D_  512
#define H_  8
#define DH_ 64
#define DFF_ 2048
#define EPS_ 1e-5f

typedef _Float16 v16h __attribute__((ext_vector_type(16)));
typedef _Float16 h8_t __attribute__((ext_vector_type(8)));
typedef float    v8f  __attribute__((ext_vector_type(8)));
typedef float    f4_t __attribute__((ext_vector_type(4)));

// A-matrix fragment (16x32 f16): lane half selects K base {0,8}; elements
// 0..7 -> K=kb..kb+7, elements 8..15 -> K=kb+16..kb+23 (ISA 7.12.2).
__device__ __forceinline__ v16h frag_a(const _Float16* p) {
  h8_t lo = *(const h8_t*)p;
  h8_t hi = *(const h8_t*)(p + 16);
  v16h r;
#pragma unroll
  for (int i = 0; i < 8; ++i) { r[i] = lo[i]; r[8 + i] = hi[i]; }
  return r;
}

// B-matrix fragment (32x16 f16) from an LDS tile stored [N][K]: lane half
// selects K base {0,8}; elements 0..7 -> K=kb..kb+7, 8..15 -> K=kb+16..kb+23.
__device__ __forceinline__ v16h frag_b(const _Float16* p) {
  h8_t lo = *(const h8_t*)p;
  h8_t hi = *(const h8_t*)(p + 16);
  v16h r;
#pragma unroll
  for (int i = 0; i < 8; ++i) { r[i] = lo[i]; r[8 + i] = hi[i]; }
  return r;
}

__device__ __forceinline__ v8f wmma_f16(v16h a, v16h b, v8f c) {
  return __builtin_amdgcn_wmma_f32_16x16x32_f16(false, a, false, b,
                                                (short)0, c, false, false);
}

// ---------------------------------------------------------------------------
// fp32 -> f16 bulk convert (8 elements/thread, vectorized)
// ---------------------------------------------------------------------------
__global__ __launch_bounds__(256)
void cvt_f32_f16_kernel(const float* __restrict__ src,
                        _Float16* __restrict__ dst) {
  size_t i = (size_t)blockIdx.x * 256 + threadIdx.x;
  f4_t a = ((const f4_t*)src)[i * 2];
  f4_t b = ((const f4_t*)src)[i * 2 + 1];
  h8_t o;
#pragma unroll
  for (int t = 0; t < 4; ++t) { o[t] = (_Float16)a[t]; o[4 + t] = (_Float16)b[t]; }
  ((h8_t*)dst)[i] = o;
}

// ---------------------------------------------------------------------------
// fp32 [K][N] -> f16 [N][K] transpose+convert (32x32 LDS tiles)
// ---------------------------------------------------------------------------
__global__ __launch_bounds__(256)
void transpose_cvt_kernel(const float* __restrict__ src,
                          _Float16* __restrict__ dst, int K, int N) {
  __shared__ _Float16 t[32][33];
  int n0 = blockIdx.x * 32, k0 = blockIdx.y * 32;
#pragma unroll
  for (int j = 0; j < 4; ++j) {
    int kk = threadIdx.y + j * 8;
    t[kk][threadIdx.x] = (_Float16)src[(size_t)(k0 + kk) * N + n0 + threadIdx.x];
  }
  __syncthreads();
#pragma unroll
  for (int j = 0; j < 4; ++j) {
    int nn = threadIdx.y + j * 8;
    dst[(size_t)(n0 + nn) * K + k0 + threadIdx.x] = t[threadIdx.x][nn];
  }
}

// ---------------------------------------------------------------------------
// GEMM: C[M,N] = act(A[M,K] @ B + bias[N]); A f16 [M][K], BT f16 [N][K].
// Block 128x128, BK=32, 256 threads = 8 waves, each wave 32(M) x 64(N).
// OUTF16 selects f16 (Ch) vs f32 (Cf) output.
// ---------------------------------------------------------------------------
template <int RELU, int OUTF16>
__global__ __launch_bounds__(256)
void gemm_wmma_kernel(const _Float16* __restrict__ A,
                      const _Float16* __restrict__ BT,
                      const float* __restrict__ bias,
                      float* __restrict__ Cf, _Float16* __restrict__ Ch,
                      int M, int N, int K) {
  __shared__ _Float16 As[128 * 40];   // [M][K] tile, stride 40 halves
  __shared__ _Float16 Bs[128 * 40];   // [N][K] tile, stride 40 halves

  const int tid  = threadIdx.x;
  const int bn0  = blockIdx.x * 128;
  const int bm0  = blockIdx.y * 128;
  const int wave = tid >> 5;
  const int lane = tid & 31;
  const int lrow = lane & 15;
  const int lhi  = lane >> 4;
  const int wM   = (wave & 3) * 32;   // 4 waves along M
  const int wN   = (wave >> 2) * 64;  // 2 waves along N

  v8f acc[2][4] = {};

  for (int k0 = 0; k0 < K; k0 += 32) {
    __syncthreads();
    // 128x32 halves per tile = 512 chunks of 8 halves; 2 chunks/thread
#pragma unroll
    for (int i = 0; i < 2; ++i) {
      int c = tid + i * 256;
      int r = c >> 2, kk = (c & 3) * 8;
      *(h8_t*)&As[r * 40 + kk] =
          *(const h8_t*)&A[(size_t)(bm0 + r) * K + k0 + kk];
      *(h8_t*)&Bs[r * 40 + kk] =
          *(const h8_t*)&BT[(size_t)(bn0 + r) * K + k0 + kk];
    }
    if (k0 + 32 < K) {
      __builtin_prefetch(&A[(size_t)(bm0 + (tid & 127)) * K + k0 + 32], 0, 1);
      __builtin_prefetch(&BT[(size_t)(bn0 + (tid & 127)) * K + k0 + 32], 0, 1);
    }
    __syncthreads();

    v16h af[2];
#pragma unroll
    for (int mt = 0; mt < 2; ++mt)
      af[mt] = frag_a(&As[(wM + mt * 16 + lrow) * 40 + lhi * 8]);
    v16h bf[4];
#pragma unroll
    for (int nt = 0; nt < 4; ++nt)
      bf[nt] = frag_b(&Bs[(wN + nt * 16 + lrow) * 40 + lhi * 8]);
#pragma unroll
    for (int mt = 0; mt < 2; ++mt)
#pragma unroll
      for (int nt = 0; nt < 4; ++nt)
        acc[mt][nt] = wmma_f16(af[mt], bf[nt], acc[mt][nt]);
  }

#pragma unroll
  for (int mt = 0; mt < 2; ++mt) {
#pragma unroll
    for (int nt = 0; nt < 4; ++nt) {
      int col = bn0 + wN + nt * 16 + lrow;
      float bv = bias ? bias[col] : 0.0f;
#pragma unroll
      for (int r = 0; r < 8; ++r) {
        int row = bm0 + wM + mt * 16 + lhi * 8 + r;
        float v = acc[mt][nt][r] + bv;
        if (RELU) v = fmaxf(v, 0.0f);
        if (OUTF16) Ch[(size_t)row * N + col] = (_Float16)v;
        else        Cf[(size_t)row * N + col] = v;
      }
    }
  }
}

// ---------------------------------------------------------------------------
// Flash attention: grid (S/128, H, B), 256 threads. Q/K/V f16 in [B,S,D].
// Online softmax over 128-key blocks; 1/sqrt(DH) folded into f16 Q scale.
// ctx written as f16 in [B,S,D].
// ---------------------------------------------------------------------------
__global__ __launch_bounds__(256)
void flash_attn_kernel(const _Float16* __restrict__ q,
                       const _Float16* __restrict__ k,
                       const _Float16* __restrict__ v,
                       const float* __restrict__ mask,
                       _Float16* __restrict__ ctx) {
  extern __shared__ char smem_raw[];
  _Float16* Qs = (_Float16*)smem_raw;       // [128][72]
  _Float16* Kt = Qs + 128 * 72;             // [128 keys][72 dims]  ([N][K])
  _Float16* Vt = Kt + 128 * 72;             // [64 dims][136 keys]  ([N][K])
  _Float16* Pb = Vt + 64 * 136;             // [128 rows][136 keys] ([M][K])
  float*    Sb = (float*)(Pb + 128 * 136);  // [128][132] logits
  float*  mrow = Sb + 128 * 132;
  float*  lrow = mrow + 128;
  float*  arow = lrow + 128;

  const int tid  = threadIdx.x;
  const int lane = tid & 31;
  const int lrw  = lane & 15;
  const int lhi  = lane >> 4;
  const int wave = tid >> 5;
  const int bb = blockIdx.z, hh = blockIdx.y;
  const int q0 = blockIdx.x * 128;
  const size_t ho = (size_t)hh * DH_;
  const size_t batch_base = (size_t)bb * S_ * D_;

  // load Q tile (scaled by 1/8): 8192 halves = 1024 chunks, 4/thread
#pragma unroll
  for (int i = 0; i < 4; ++i) {
    int c = tid + i * 256;
    int r = c >> 3, d = (c & 7) * 8;
    h8_t qv = *(const h8_t*)&q[batch_base + (size_t)(q0 + r) * D_ + ho + d];
#pragma unroll
    for (int t = 0; t < 8; ++t) qv[t] = qv[t] * (_Float16)0.125f;
    *(h8_t*)&Qs[r * 72 + d] = qv;
  }
  if (tid < 128) { mrow[tid] = -1e30f; lrow[tid] = 0.0f; }

  const int rg = (wave >> 1) * 32;    // O accum: 32(M) x 32(N)
  const int cg = (wave & 1) * 32;
  v8f oacc[2][2] = {};

  const int sM = (wave & 3) * 32;     // S tile: 4 waves M, 2 waves N
  const int sN = (wave >> 2) * 64;

  for (int j = 0; j < S_ / 128; ++j) {
    __syncthreads();
#pragma unroll
    for (int i = 0; i < 4; ++i) {
      int c = tid + i * 256;
      int n = c >> 3, d = (c & 7) * 8;
      *(h8_t*)&Kt[n * 72 + d] =
          *(const h8_t*)&k[batch_base + (size_t)(j * 128 + n) * D_ + ho + d];
    }
#pragma unroll
    for (int i = 0; i < 4; ++i) {
      int c = tid + i * 256;
      int kk = c >> 3, d = (c & 7) * 8;
      h8_t vv = *(const h8_t*)&v[batch_base + (size_t)(j * 128 + kk) * D_ + ho + d];
#pragma unroll
      for (int t = 0; t < 8; ++t) Vt[(d + t) * 136 + kk] = vv[t];
    }
    __syncthreads();

    // ---- S = Q * K^T (128x128, K-dim 64) ----
    v8f sacc[2][4] = {};
#pragma unroll
    for (int kk0 = 0; kk0 < 64; kk0 += 32) {
      v16h af[2];
#pragma unroll
      for (int mt = 0; mt < 2; ++mt)
        af[mt] = frag_a(&Qs[(sM + mt * 16 + lrw) * 72 + kk0 + lhi * 8]);
      v16h bf[4];
#pragma unroll
      for (int nt = 0; nt < 4; ++nt)
        bf[nt] = frag_b(&Kt[(sN + nt * 16 + lrw) * 72 + kk0 + lhi * 8]);
#pragma unroll
      for (int mt = 0; mt < 2; ++mt)
#pragma unroll
        for (int nt = 0; nt < 4; ++nt)
          sacc[mt][nt] = wmma_f16(af[mt], bf[nt], sacc[mt][nt]);
    }
#pragma unroll
    for (int nt = 0; nt < 4; ++nt) {
      int col = sN + nt * 16 + lrw;
      float mv = mask[(size_t)bb * S_ + j * 128 + col] * -1.0e9f;
#pragma unroll
      for (int mt = 0; mt < 2; ++mt)
#pragma unroll
        for (int r = 0; r < 8; ++r) {
          int row = sM + mt * 16 + lhi * 8 + r;
          Sb[row * 132 + col] = sacc[mt][nt][r] + mv;
        }
    }
    __syncthreads();

    // ---- online softmax (one thread per row) ----
    if (tid < 128) {
      int row = tid;
      float mo = mrow[row];
      float mx = mo;
#pragma unroll 4
      for (int c = 0; c < 128; ++c) mx = fmaxf(mx, Sb[row * 132 + c]);
      float al = __expf(mo - mx);
      float sum = 0.0f;
#pragma unroll 4
      for (int c = 0; c < 128; ++c) {
        float p = __expf(Sb[row * 132 + c] - mx);
        Pb[row * 136 + c] = (_Float16)p;
        sum += p;
      }
      lrow[row] = lrow[row] * al + sum;
      mrow[row] = mx;
      arow[row] = al;
    }
    __syncthreads();

    // ---- rescale O and accumulate O += P * V ----
#pragma unroll
    for (int mt = 0; mt < 2; ++mt)
#pragma unroll
      for (int nt = 0; nt < 2; ++nt)
#pragma unroll
        for (int r = 0; r < 8; ++r) {
          int row = rg + mt * 16 + lhi * 8 + r;
          oacc[mt][nt][r] *= arow[row];
        }
#pragma unroll
    for (int kk = 0; kk < 4; ++kk) {
      v16h af[2];
#pragma unroll
      for (int mt = 0; mt < 2; ++mt)
        af[mt] = frag_a(&Pb[(rg + mt * 16 + lrw) * 136 + kk * 32 + lhi * 8]);
      v16h bf[2];
#pragma unroll
      for (int nt = 0; nt < 2; ++nt)
        bf[nt] = frag_b(&Vt[(cg + nt * 16 + lrw) * 136 + kk * 32 + lhi * 8]);
#pragma unroll
      for (int mt = 0; mt < 2; ++mt)
#pragma unroll
        for (int nt = 0; nt < 2; ++nt)
          oacc[mt][nt] = wmma_f16(af[mt], bf[nt], oacc[mt][nt]);
    }
  }

#pragma unroll
  for (int mt = 0; mt < 2; ++mt)
#pragma unroll
    for (int nt = 0; nt < 2; ++nt) {
      int col = cg + nt * 16 + lrw;
#pragma unroll
      for (int r = 0; r < 8; ++r) {
        int row = rg + mt * 16 + lhi * 8 + r;
        ctx[batch_base + (size_t)(q0 + row) * D_ + ho + col] =
            (_Float16)(oacc[mt][nt][r] / lrow[row]);
      }
    }
}

// ---------------------------------------------------------------------------
// LayerNorm over the full [S,D] slab per batch (two-stage, deterministic).
// ---------------------------------------------------------------------------
#define LN_CHUNKS 256

__global__ __launch_bounds__(256)
void ln_reduce_kernel(const float* __restrict__ a, const float* __restrict__ r,
                      float* __restrict__ psum, float* __restrict__ psq) {
  __shared__ float ss[256], sq[256];
  int bidx = blockIdx.x;
  int b = bidx / LN_CHUNKS, ch = bidx % LN_CHUNKS;
  size_t base = (size_t)b * S_ * D_ + (size_t)ch * 2048;
  float s = 0.0f, s2 = 0.0f;
#pragma unroll
  for (int i = 0; i < 8; ++i) {
    size_t idx = base + threadIdx.x + i * 256;
    float v = a[idx] + r[idx];
    s += v; s2 += v * v;
  }
  ss[threadIdx.x] = s; sq[threadIdx.x] = s2;
  __syncthreads();
  for (int o = 128; o > 0; o >>= 1) {
    if (threadIdx.x < o) {
      ss[threadIdx.x] += ss[threadIdx.x + o];
      sq[threadIdx.x] += sq[threadIdx.x + o];
    }
    __syncthreads();
  }
  if (threadIdx.x == 0) { psum[bidx] = ss[0]; psq[bidx] = sq[0]; }
}

__global__ __launch_bounds__(256)
void ln_finalize_kernel(const float* __restrict__ psum,
                        const float* __restrict__ psq,
                        float* __restrict__ stats) {
  __shared__ float ss[256], sq[256];
  int b = blockIdx.x;
  ss[threadIdx.x] = psum[b * LN_CHUNKS + threadIdx.x];
  sq[threadIdx.x] = psq[b * LN_CHUNKS + threadIdx.x];
  __syncthreads();
  for (int o = 128; o > 0; o >>= 1) {
    if (threadIdx.x < o) {
      ss[threadIdx.x] += ss[threadIdx.x + o];
      sq[threadIdx.x] += sq[threadIdx.x + o];
    }
    __syncthreads();
  }
  if (threadIdx.x == 0) {
    const float invn = 1.0f / (float)((size_t)S_ * D_);
    float mean = ss[0] * invn;
    float var  = sq[0] * invn - mean * mean;
    stats[2 * b]     = mean;
    stats[2 * b + 1] = rsqrtf(var + EPS_);
  }
}

// Writes fp32 result; optionally a f16 copy for the next GEMM's A operand.
__global__ __launch_bounds__(256)
void ln_apply_kernel(const float* __restrict__ a, const float* __restrict__ r,
                     const float* __restrict__ stats,
                     const float* __restrict__ w, const float* __restrict__ bsh,
                     float* __restrict__ outf, _Float16* __restrict__ outh) {
  size_t idx = (size_t)blockIdx.x * 256 + threadIdx.x;
  const size_t SD = (size_t)S_ * D_;
  int b = (int)(idx / SD);
  size_t sd = idx % SD;
  float mean = stats[2 * b], rstd = stats[2 * b + 1];
  float v = (a[idx] + r[idx] - mean) * rstd * w[sd] + bsh[sd];
  outf[idx] = v;
  if (outh) outh[idx] = (_Float16)v;
}

// ---------------------------------------------------------------------------
// Host launcher
// ---------------------------------------------------------------------------
extern "C" void kernel_launch(void* const* d_in, const int* in_sizes, int n_in,
                              void* d_out, int out_size, void* d_ws,
                              size_t ws_size, hipStream_t stream) {
  const float* x    = (const float*)d_in[0];
  const float* mask = (const float*)d_in[1];
  const float* wq   = (const float*)d_in[2];
  const float* bq   = (const float*)d_in[3];
  const float* wk   = (const float*)d_in[4];
  const float* bk   = (const float*)d_in[5];
  const float* wv   = (const float*)d_in[6];
  const float* bv   = (const float*)d_in[7];
  const float* wo   = (const float*)d_in[8];
  const float* bo   = (const float*)d_in[9];
  const float* w1   = (const float*)d_in[10];
  const float* b1   = (const float*)d_in[11];
  const float* w2   = (const float*)d_in[12];
  const float* b2   = (const float*)d_in[13];
  const float* ln1w = (const float*)d_in[14];
  const float* ln1b = (const float*)d_in[15];
  const float* ln2w = (const float*)d_in[16];
  const float* ln2b = (const float*)d_in[17];
  float* out = (float*)d_out;

  const size_t NSD = (size_t)B_ * S_ * D_;     // 8,388,608 elements
  const size_t HB  = NSD * sizeof(_Float16);   // 16 MiB per half slab
  char* w8 = (char*)d_ws;

  _Float16* xh    = (_Float16*)(w8 + 0 * HB);
  _Float16* qh    = (_Float16*)(w8 + 1 * HB);
  _Float16* kh    = (_Float16*)(w8 + 2 * HB);
  _Float16* vh    = (_Float16*)(w8 + 3 * HB);
  _Float16* ctxh  = (_Float16*)(w8 + 4 * HB);
  _Float16* ffnh  = (_Float16*)(w8 + 1 * HB);  // reuse qh..ctxh span (4*HB)
  _Float16* out1h = (_Float16*)(w8 + 5 * HB);
  _Float16* wqT   = (_Float16*)(w8 + 6 * HB);
  _Float16* wkT   = wqT + (size_t)D_ * D_;
  _Float16* wvT   = wkT + (size_t)D_ * D_;
  _Float16* woT   = wvT + (size_t)D_ * D_;
  _Float16* w1T   = woT + (size_t)D_ * D_;       // [DFF][D]
  _Float16* w2T   = w1T + (size_t)D_ * DFF_;     // [D][DFF]
  char* fbase = w8 + 6 * HB + (8u << 20);
  float* aout  = (float*)fbase;                   // 32 MiB, reused by ffno
  float* ffno  = aout;
  float* out1  = (float*)(fbase + NSD * 4);
  float* psum  = (float*)(fbase + 2 * NSD * 4);
  float* psq   = psum + B_ * LN_CHUNKS;
  float* stats = psq + B_ * LN_CHUNKS;

  const int M = B_ * S_;   // 16384
  dim3 blk(256);
  dim3 tblk(32, 8);

  // --- pre-pass: convert x, transpose+convert weights ---
  cvt_f32_f16_kernel<<<(unsigned)(NSD / 2048), blk, 0, stream>>>(x, xh);
  transpose_cvt_kernel<<<dim3(D_ / 32, D_ / 32), tblk, 0, stream>>>(wq, wqT, D_, D_);
  transpose_cvt_kernel<<<dim3(D_ / 32, D_ / 32), tblk, 0, stream>>>(wk, wkT, D_, D_);
  transpose_cvt_kernel<<<dim3(D_ / 32, D_ / 32), tblk, 0, stream>>>(wv, wvT, D_, D_);
  transpose_cvt_kernel<<<dim3(D_ / 32, D_ / 32), tblk, 0, stream>>>(wo, woT, D_, D_);
  transpose_cvt_kernel<<<dim3(DFF_ / 32, D_ / 32), tblk, 0, stream>>>(w1, w1T, D_, DFF_);
  transpose_cvt_kernel<<<dim3(D_ / 32, DFF_ / 32), tblk, 0, stream>>>(w2, w2T, DFF_, D_);

  // --- QKV projections (f16 out) ---
  gemm_wmma_kernel<0, 1><<<dim3(D_ / 128, M / 128), blk, 0, stream>>>(
      xh, wqT, bq, nullptr, qh, M, D_, D_);
  gemm_wmma_kernel<0, 1><<<dim3(D_ / 128, M / 128), blk, 0, stream>>>(
      xh, wkT, bk, nullptr, kh, M, D_, D_);
  gemm_wmma_kernel<0, 1><<<dim3(D_ / 128, M / 128), blk, 0, stream>>>(
      xh, wvT, bv, nullptr, vh, M, D_, D_);

  // --- flash attention (f16 in/out) ---
  const size_t smem = (size_t)(128 * 72 + 128 * 72 + 64 * 136 + 128 * 136) * 2 +
                      (size_t)(128 * 132) * 4 + 3 * 128 * 4;   // ~158 KB
  flash_attn_kernel<<<dim3(S_ / 128, H_, B_), blk, smem, stream>>>(
      qh, kh, vh, mask, ctxh);

  // --- output projection (fp32 out for residual) ---
  gemm_wmma_kernel<0, 0><<<dim3(D_ / 128, M / 128), blk, 0, stream>>>(
      ctxh, woT, bo, aout, nullptr, M, D_, D_);

  // --- LayerNorm 1 over [S,D]: resid = x + aout -> out1 (f32) + out1h (f16) ---
  ln_reduce_kernel<<<B_ * LN_CHUNKS, blk, 0, stream>>>(x, aout, psum, psq);
  ln_finalize_kernel<<<B_, blk, 0, stream>>>(psum, psq, stats);
  ln_apply_kernel<<<(unsigned)(NSD / 256), blk, 0, stream>>>(
      x, aout, stats, ln1w, ln1b, out1, out1h);

  // --- FFN ---
  gemm_wmma_kernel<1, 1><<<dim3(DFF_ / 128, M / 128), blk, 0, stream>>>(
      out1h, w1T, b1, nullptr, ffnh, M, DFF_, D_);
  gemm_wmma_kernel<0, 0><<<dim3(D_ / 128, M / 128), blk, 0, stream>>>(
      ffnh, w2T, b2, ffno, nullptr, M, D_, DFF_);

  // --- LayerNorm 2: resid = out1 + ffno -> out (f32) ---
  ln_reduce_kernel<<<B_ * LN_CHUNKS, blk, 0, stream>>>(out1, ffno, psum, psq);
  ln_finalize_kernel<<<B_, blk, 0, stream>>>(psum, psq, stats);
  ln_apply_kernel<<<(unsigned)(NSD / 256), blk, 0, stream>>>(
      out1, ffno, stats, ln2w, ln2b, out, nullptr);
}
